// CustomAttention_38508676776099
// MI455X (gfx1250) — matrix operand order repaired and checked by reading, not compiled, not run on hardware
//
#include <hip/hip_runtime.h>

// ---------------------------------------------------------------------------
// CustomAttention for MI455X (gfx1250): all GEMMs via v_wmma_f32_16x16x32_bf16,
// double-buffered LDS tiles, async global->LDS staging (ASYNCcnt) for bf16
// sources when the builtin is available. f32 sources are converted to bf16 in
// VALU during staging; accumulation is f32 in WMMA accumulators.
// Pipeline:
//   1) q/k/v = proj(x, W, b)           (bf16 out)
//   2) scores = (q^T)(k^T) * 1/32      (f32 out)
//   3) attn = softmax(scores)          (bf16 out)
//   4) o = attn @ v                    (bf16 out)
//   5) out = o @ Wo^T + bo             (f32 out -> d_out)
// ---------------------------------------------------------------------------

#ifndef USE_ASYNC_LDS
#  if defined(__has_builtin)
#    if __has_builtin(__builtin_amdgcn_global_load_async_to_lds_b128)
#      define USE_ASYNC_LDS 1
#    else
#      define USE_ASYNC_LDS 0
#    endif
#  else
#    define USE_ASYNC_LDS 0
#  endif
#endif

typedef __attribute__((ext_vector_type(16))) __bf16 v16bf;
typedef __attribute__((ext_vector_type(8)))  float  v8f;
typedef int async_v4i __attribute__((vector_size(16)));   // matches builtin param

static __device__ __forceinline__ unsigned short f32_to_bf16(float f) {
    unsigned int u = __builtin_bit_cast(unsigned int, f);
    u += 0x7FFFu + ((u >> 16) & 1u);          // round to nearest even
    return (unsigned short)(u >> 16);
}

union Frag32B {
    unsigned short s[16];
    uint4          u[2];
    v16bf          v;
};

// load 16 source elements (f32 -> convert, bf16 -> copy) into a 32-byte frag
static __device__ __forceinline__ void load16(const float* p, Frag32B& f) {
    const float4* p4 = reinterpret_cast<const float4*>(p);
#pragma unroll
    for (int i = 0; i < 4; ++i) {
        float4 x = p4[i];
        f.s[i * 4 + 0] = f32_to_bf16(x.x);
        f.s[i * 4 + 1] = f32_to_bf16(x.y);
        f.s[i * 4 + 2] = f32_to_bf16(x.z);
        f.s[i * 4 + 3] = f32_to_bf16(x.w);
    }
}
static __device__ __forceinline__ void load16(const unsigned short* p, Frag32B& f) {
    const uint4* p4 = reinterpret_cast<const uint4*>(p);
    f.u[0] = p4[0];
    f.u[1] = p4[1];
}

// direct (K-contiguous) staging of 16 elements: f32 converts through VGPRs,
// bf16 uses GLOBAL_LOAD_ASYNC_TO_LDS_B128 when available (inst offset applies
// to both the global and LDS address, so two calls with offset 0 / 16 move
// the full 32 bytes without VGPR data movement).
static __device__ __forceinline__ void stage16_direct(const float* src, unsigned short* dst) {
    Frag32B f;
    load16(src, f);
    *reinterpret_cast<uint4*>(dst)     = f.u[0];
    *reinterpret_cast<uint4*>(dst + 8) = f.u[1];
}
static __device__ __forceinline__ void stage16_direct(const unsigned short* src, unsigned short* dst) {
#if USE_ASYNC_LDS
    __attribute__((address_space(1))) async_v4i* g =
        (__attribute__((address_space(1))) async_v4i*)src;
    __attribute__((address_space(3))) async_v4i* l =
        (__attribute__((address_space(3))) async_v4i*)dst;
    __builtin_amdgcn_global_load_async_to_lds_b128(g, l, 0, 0);
    __builtin_amdgcn_global_load_async_to_lds_b128(g, l, 16, 0);
#else
    Frag32B f;
    load16(src, f);
    *reinterpret_cast<uint4*>(dst)     = f.u[0];
    *reinterpret_cast<uint4*>(dst + 8) = f.u[1];
#endif
}

static __device__ __forceinline__ void storeC(float* p, float v) { *p = v; }
static __device__ __forceinline__ void storeC(unsigned short* p, float v) { *p = f32_to_bf16(v); }

// ---------------------------------------------------------------------------
// Generic tiled WMMA GEMM:  C[m,n] = scale * sum_k Aval[m,k]*Bval[k,n] (+bias[n])
//   TRA = false: A source is [M,K] row-major (K contiguous)
//   TRA = true : A source is [K,M] row-major (transpose-staged into LDS)
//   TRB = false: B source is [N,K] row-major (K contiguous; weight layout)
//   TRB = true : B source is [K,N] row-major (transpose-staged into LDS)
// Block: 256 threads = 8 waves; tile 128x128, K-step 32, double-buffered LDS.
// Wave grid 4(M)x2(N): each wave owns 32 rows x 64 cols -> per K-step:
// 2 A-frags + 4 B-frags feed 8 WMMAs (each B-frag reused by 2 WMMAs).
// ---------------------------------------------------------------------------
constexpr int BLK_M = 128;
constexpr int BLK_N = 128;
constexpr int BLK_K = 32;
constexpr int PITCH = BLK_K + 8;   // 40 ushorts = 80B rows: 16B-aligned, bank-spread

template <typename AT, typename BT, typename CT, bool TRA, bool TRB, bool BIAS>
__global__ __launch_bounds__(256, 2)
void gemm_bf16_wmma(const AT* __restrict__ A, const BT* __restrict__ B,
                    CT* __restrict__ C, const float* __restrict__ bias,
                    float scale, int M, int N, int K,
                    int lda, int ldb, int ldc,
                    long long sA, long long sB, long long sC) {
    __shared__ __align__(16) unsigned short As[2][BLK_M * PITCH];
    __shared__ __align__(16) unsigned short Bs[2][BLK_N * PITCH];

    const int tid  = threadIdx.x;
    const int lane = tid & 31;
    const int wave = tid >> 5;
    const int n0   = blockIdx.x * BLK_N;
    const int m0   = blockIdx.y * BLK_M;

    const AT* Ab = A + (long long)blockIdx.z * sA;
    const BT* Bb = B + (long long)blockIdx.z * sB;
    CT*       Cb = C + (long long)blockIdx.z * sC;

    // ASYNCcnt is used only by the bf16 direct-staging paths
    constexpr bool ASYNC_USED =
        USE_ASYNC_LDS && ((!TRA && sizeof(AT) == 2) || (!TRB && sizeof(BT) == 2));

    v8f acc0[4], acc1[4];
#pragma unroll
    for (int j = 0; j < 4; ++j)
#pragma unroll
        for (int r = 0; r < 8; ++r) { acc0[j][r] = 0.0f; acc1[j][r] = 0.0f; }

    // staging maps: direct (128 rows x 32 K, 16 elems/thread) and transposed
    const int dr = tid >> 1;          // 0..127
    const int dk = (tid & 1) * 16;    // 0 or 16
    const int tk = tid >> 3;          // 0..31   (K index)
    const int tc = (tid & 7) * 16;    // 0..112  (row-in-tile base)

    auto stage_tiles = [&](int buf, int k0) {
        // ---- A tile -> As[buf][m][k] ----
        if (!TRA) {
            stage16_direct(Ab + (long long)(m0 + dr) * lda + (k0 + dk),
                           &As[buf][dr * PITCH + dk]);
        } else {
            Frag32B fa;
            load16(Ab + (long long)(k0 + tk) * lda + (m0 + tc), fa);
#pragma unroll
            for (int j = 0; j < 16; ++j) As[buf][(tc + j) * PITCH + tk] = fa.s[j];
        }
        // ---- B tile -> Bs[buf][n][k] ----
        if (!TRB) {
            stage16_direct(Bb + (long long)(n0 + dr) * ldb + (k0 + dk),
                           &Bs[buf][dr * PITCH + dk]);
        } else {
            Frag32B fb;
            load16(Bb + (long long)(k0 + tk) * ldb + (n0 + tc), fb);
#pragma unroll
            for (int j = 0; j < 16; ++j) Bs[buf][(tc + j) * PITCH + tk] = fb.s[j];
        }
    };

    stage_tiles(0, 0);

    // wave -> 32x64 sub-tile: row group wr (0..3), col group wc (0..1)
    const int wr     = wave >> 1;
    const int wc     = wave & 1;
    const int lane15 = lane & 15;
    // A 16x32 bf16 frag: lanes 0-15 row=lane  K{0..7,16..23}
    //                    lanes16-31 row=l-16  K{8..15,24..31}
    const int mrowA0 = wr * 32 + lane15;
    const int mrowA1 = mrowA0 + 16;
    const int ka     = (lane < 16) ? 0 : 8;
    // B 32x16 bf16 frag: lanes 0-15 col=lane K{0..7,16..23}; lanes16-31 col=l-16 K{8..15,24..31}
    const int kb     = (lane < 16) ? 0 : 8;

    int cur = 0;
    for (int k0 = 0; k0 < K; k0 += BLK_K, cur ^= 1) {
        if (ASYNC_USED) asm volatile("s_wait_asynccnt 0" ::: "memory");
        __syncthreads();                       // publish tile `cur`
        if (k0 + BLK_K < K) stage_tiles(cur ^ 1, k0 + BLK_K);   // overlap next tile

        Frag32B af0, af1;
        af0.u[0] = *reinterpret_cast<const uint4*>(&As[cur][mrowA0 * PITCH + ka]);
        af0.u[1] = *reinterpret_cast<const uint4*>(&As[cur][mrowA0 * PITCH + ka + 16]);
        af1.u[0] = *reinterpret_cast<const uint4*>(&As[cur][mrowA1 * PITCH + ka]);
        af1.u[1] = *reinterpret_cast<const uint4*>(&As[cur][mrowA1 * PITCH + ka + 16]);

#pragma unroll
        for (int j = 0; j < 4; ++j) {
            const int ncol = wc * 64 + j * 16 + lane15;
            Frag32B bfr;
            bfr.u[0] = *reinterpret_cast<const uint4*>(&Bs[cur][ncol * PITCH + kb]);
            bfr.u[1] = *reinterpret_cast<const uint4*>(&Bs[cur][ncol * PITCH + kb + 16]);
            acc0[j] = __builtin_amdgcn_wmma_f32_16x16x32_bf16(
                false, af0.v, false, bfr.v, (short)0, acc0[j], false, false);
            acc1[j] = __builtin_amdgcn_wmma_f32_16x16x32_bf16(
                false, af1.v, false, bfr.v, (short)0, acc1[j], false, false);
        }
    }

    // ---- epilogue: C 16x16 f32 layout: VGPR r -> M=r (lanes<16) / M=8+r ----
#pragma unroll
    for (int j = 0; j < 4; ++j) {
        const int col = n0 + wc * 64 + j * 16 + lane15;
        const float bv = BIAS ? bias[col] : 0.0f;
#pragma unroll
        for (int r = 0; r < 8; ++r) {
            const int mr = wr * 32 + ((lane < 16) ? r : (8 + r));
            storeC(&Cb[(long long)(m0 + mr) * ldc + col],      acc0[j][r] * scale + bv);
            storeC(&Cb[(long long)(m0 + mr + 16) * ldc + col], acc1[j][r] * scale + bv);
        }
    }
    (void)M; (void)N;
}

// ---------------------------------------------------------------------------
// Row softmax: one 256-thread block per row of 1024 f32; bf16 output.
// ---------------------------------------------------------------------------
__global__ __launch_bounds__(256)
void softmax_rows_kernel(const float* __restrict__ in, unsigned short* __restrict__ out,
                         int ncols) {
    __shared__ float redmax[8];
    __shared__ float redsum[8];
    const long long row = blockIdx.x;
    const int tid = threadIdx.x;
    const float4 x = reinterpret_cast<const float4*>(in + row * ncols)[tid];

    float m = fmaxf(fmaxf(x.x, x.y), fmaxf(x.z, x.w));
#pragma unroll
    for (int off = 16; off > 0; off >>= 1) m = fmaxf(m, __shfl_xor(m, off, 32));
    if ((tid & 31) == 0) redmax[tid >> 5] = m;
    __syncthreads();
    float rm = redmax[0];
#pragma unroll
    for (int i = 1; i < 8; ++i) rm = fmaxf(rm, redmax[i]);

    const float e0 = __expf(x.x - rm), e1 = __expf(x.y - rm);
    const float e2 = __expf(x.z - rm), e3 = __expf(x.w - rm);
    float s = e0 + e1 + e2 + e3;
#pragma unroll
    for (int off = 16; off > 0; off >>= 1) s += __shfl_xor(s, off, 32);
    if ((tid & 31) == 0) redsum[tid >> 5] = s;
    __syncthreads();
    float rs = 0.0f;
#pragma unroll
    for (int i = 0; i < 8; ++i) rs += redsum[i];
    const float inv = 1.0f / rs;

    uint2 o;
    o.x = (unsigned int)f32_to_bf16(e0 * inv) | ((unsigned int)f32_to_bf16(e1 * inv) << 16);
    o.y = (unsigned int)f32_to_bf16(e2 * inv) | ((unsigned int)f32_to_bf16(e3 * inv) << 16);
    reinterpret_cast<uint2*>(out + row * ncols)[tid] = o;
}

// ---------------------------------------------------------------------------
extern "C" void kernel_launch(void* const* d_in, const int* in_sizes, int n_in,
                              void* d_out, int out_size, void* d_ws, size_t ws_size,
                              hipStream_t stream) {
    const int B = 16, T = 1024, S = 1024, E = 1024;
    const float* query = (const float*)d_in[0];
    const float* key_  = (const float*)d_in[1];
    const float* value = (const float*)d_in[2];
    const float* Wq = (const float*)d_in[3]; const float* bq = (const float*)d_in[4];
    const float* Wk = (const float*)d_in[5]; const float* bk = (const float*)d_in[6];
    const float* Wv = (const float*)d_in[7]; const float* bv = (const float*)d_in[8];
    const float* Wo = (const float*)d_in[9]; const float* bo = (const float*)d_in[10];
    float* out = (float*)d_out;

    const long long NB = (long long)B * T * E;   // 16M elements
    char* ws = (char*)d_ws;
    unsigned short* qb     = (unsigned short*)(ws);            // bf16 [B,T,E]
    unsigned short* kb     = (unsigned short*)(ws + 2 * NB);   // bf16 [B,S,E]
    unsigned short* vb     = (unsigned short*)(ws + 4 * NB);   // bf16 [B,S,E]
    unsigned short* attn   = (unsigned short*)(ws + 6 * NB);   // bf16 [B,E,S]
    unsigned short* ob     = (unsigned short*)(ws + 8 * NB);   // bf16 [B,E,E]
    float*          scores = (float*)(ws + 10 * NB);           // f32  [B,E,S]

    const dim3 blk(256);
    const float scale = 0.03125f;  // 1/sqrt(1024)

    // 1) projections: [B*T,E] @ W^T + b -> bf16
    {
        dim3 grd(E / 128, (B * T) / 128, 1);
        gemm_bf16_wmma<float, float, unsigned short, false, false, true>
            <<<grd, blk, 0, stream>>>(query, Wq, qb, bq, 1.0f, B * T, E, E, E, E, E, 0, 0, 0);
        gemm_bf16_wmma<float, float, unsigned short, false, false, true>
            <<<grd, blk, 0, stream>>>(key_, Wk, kb, bk, 1.0f, B * S, E, E, E, E, E, 0, 0, 0);
        gemm_bf16_wmma<float, float, unsigned short, false, false, true>
            <<<grd, blk, 0, stream>>>(value, Wv, vb, bv, 1.0f, B * S, E, E, E, E, E, 0, 0, 0);
    }
    // 2) scores[b,i,s] = scale * sum_t q[b,t,i] * k[b,s,t]
    {
        dim3 grd(S / 128, E / 128, B);
        gemm_bf16_wmma<unsigned short, unsigned short, float, true, false, false>
            <<<grd, blk, 0, stream>>>(qb, kb, scores, nullptr, scale, E, S, T,
                                      E, T, S,
                                      (long long)T * E, (long long)S * T, (long long)E * S);
    }
    // 3) softmax over s -> bf16
    softmax_rows_kernel<<<dim3(B * E), blk, 0, stream>>>(scores, attn, S);
    // 4) o[b,i,e] = sum_s attn[b,i,s] * v[b,s,e]
    {
        dim3 grd(E / 128, E / 128, B);
        gemm_bf16_wmma<unsigned short, unsigned short, unsigned short, false, true, false>
            <<<grd, blk, 0, stream>>>(attn, vb, ob, nullptr, 1.0f, E, E, S,
                                      S, E, E,
                                      (long long)E * S, (long long)S * E, (long long)E * E);
    }
    // 5) out = o @ Wo^T + bo -> f32
    {
        dim3 grd(E / 128, (B * E) / 128, 1);
        gemm_bf16_wmma<unsigned short, float, float, false, false, true>
            <<<grd, blk, 0, stream>>>(ob, Wo, out, bo, 1.0f, B * E, E, E, E, E, E, 0, 0, 0);
    }
    (void)in_sizes; (void)n_in; (void)out_size; (void)ws_size;
}
